// SGAE_40699110097041
// MI455X (gfx1250) — matrix operand order repaired and checked by reading, not compiled, not run on hardware
//
#include <hip/hip_runtime.h>
#include <hip/hip_bf16.h>
#include <math.h>

// ---------------------------------------------------------------------------
// SGAE forward on MI455X (gfx1250, wave32, WMMA).
//   z = relu(X@W) @ W2          -> fused encode kernel, bf16 WMMA / f32 acc
//   out = sigmoid(rowdot(z[e0], z[e1])) for train edges then false edges
// Encode is HBM-bound on X (205 MB ~ 8.8us @ 23.3 TB/s); bf16 16x16x32 WMMA
// keeps compute under the memory roofline (fp32 16x16x4 WMMA would not).
// Decode gathers hit L2 (z = 25.6 MB << 192 MB L2).
// This revision preloads all B fragments into distinct registers so the
// ds_load_b128s issue as a clause and the 8 WMMAs run back-to-back instead of
// each one stalling on s_wait_dscnt 0.
// ---------------------------------------------------------------------------

typedef __bf16 bf16;
typedef __attribute__((ext_vector_type(16))) __bf16 v16bf;
typedef __attribute__((ext_vector_type(8)))  __bf16 v8bf;
typedef __attribute__((ext_vector_type(8)))  float  v8f;

#define K1 512   // input_dim
#define N1 128   // hidden1
#define N2 64    // z_dim

// 8 waves/block, wave w computes rows [blk*128 + w*16, +16).
__global__ __launch_bounds__(256) void sgae_encode_kernel(
    const float* __restrict__ X, const float* __restrict__ W,
    const float* __restrict__ W2, float* __restrict__ Z, int n_nodes)
{
    // B fragments of current 32xN1 W tile, pre-swizzled to WMMA layout (8 KB)
    __shared__ __align__(32) bf16 sW[32 * N1];
    // All B fragments of W2 (4 ksteps x 4 ntiles), pre-swizzled (16 KB)
    __shared__ __align__(32) bf16 sW2[128 * N2];
    // Per-wave h strip (16 x 128 bf16), for C-layout -> A-layout shuffle (32 KB)
    __shared__ __align__(32) bf16 sH[8][16 * N1];

    const int tid   = threadIdx.x;
    const int wave  = tid >> 5;
    const int lane  = tid & 31;
    const int lhalf = lane >> 4;   // 0: lanes 0-15, 1: lanes 16-31
    const int lmod  = lane & 15;
    const int koff  = lhalf * 8;   // A-fragment K sub-offset per ISA layout

    // ---- stage W2 fragments once -------------------------------------------
    // frag index: (((k2*4 + ntile)*32) + l)*16 + j ;
    // element K = k2*32 + (j>=8?16:0) + (l>=16?8:0) + (j&7), N = ntile*16 + (l&15)
    for (int idx = tid; idx < 128 * N2; idx += 256) {
        int grp = idx >> 9;            // k2step*4 + ntile (0..15)
        int l   = (idx >> 4) & 31;
        int j   = idx & 15;
        int k   = (grp >> 2) * 32 + ((j >> 3) << 4) + ((l >> 4) << 3) + (j & 7);
        int n   = ((grp & 3) << 4) + (l & 15);
        sW2[idx] = (bf16)W2[k * N2 + n];
    }

    const int rowBase = blockIdx.x * 128 + wave * 16;
    int rowA = rowBase + lmod;
    if (rowA > n_nodes - 1) rowA = n_nodes - 1;   // clamp: keep EXEC all-1s
    const float* xr = X + (size_t)rowA * K1;

    v8f acc[8];
    #pragma unroll
    for (int t = 0; t < 8; ++t)
        #pragma unroll
        for (int i = 0; i < 8; ++i) acc[t][i] = 0.0f;

    // ---- GEMM1: h = X @ W (K = 512, 16 K-steps of 32) ----------------------
    for (int kb = 0; kb < K1; kb += 32) {
        // stage W[kb:kb+32, 0:128] as 8 B-fragments (ntile 0..7)
        for (int idx = tid; idx < 32 * N1; idx += 256) {
            int ntile = idx >> 9;          // 0..7
            int l     = (idx >> 4) & 31;
            int j     = idx & 15;
            int kl    = ((j >> 3) << 4) + ((l >> 4) << 3) + (j & 7);  // (j>=8?16:0)+(l>=16?8:0)+(j&7)
            int n     = (ntile << 4) + (l & 15);
            sW[idx]   = (bf16)W[(kb + kl) * N1 + n];
        }
        __syncthreads();

        if (kb + 32 < K1) __builtin_prefetch(xr + kb + 32, 0, 1);

        // A fragment: row = lmod, lanes<16 hold K {0..7,16..23}, lanes>=16 {8..15,24..31}
        float4 f0 = *(const float4*)(xr + kb + koff);
        float4 f1 = *(const float4*)(xr + kb + koff + 4);
        float4 f2 = *(const float4*)(xr + kb + 16 + koff);
        float4 f3 = *(const float4*)(xr + kb + 16 + koff + 4);
        v16bf a;
        a[0]=(bf16)f0.x; a[1]=(bf16)f0.y; a[2]=(bf16)f0.z; a[3]=(bf16)f0.w;
        a[4]=(bf16)f1.x; a[5]=(bf16)f1.y; a[6]=(bf16)f1.z; a[7]=(bf16)f1.w;
        a[8]=(bf16)f2.x; a[9]=(bf16)f2.y; a[10]=(bf16)f2.z; a[11]=(bf16)f2.w;
        a[12]=(bf16)f3.x; a[13]=(bf16)f3.y; a[14]=(bf16)f3.z; a[15]=(bf16)f3.w;

        // Preload all 8 B fragments (distinct regs -> one ds clause), then
        // run the 8 WMMAs back-to-back.
        v16bf bfrag[8];
        #pragma unroll
        for (int t = 0; t < 8; ++t)
            bfrag[t] = *(const v16bf*)&sW[(t * 32 + lane) * 16];
        #pragma unroll
        for (int t = 0; t < 8; ++t)
            acc[t] = __builtin_amdgcn_wmma_f32_16x16x32_bf16(
                false, a, false, bfrag[t], (short)0, acc[t], false, false);
        __syncthreads();
    }

    // ---- ReLU + stage h strip as bf16 in LDS (C layout -> [m][n]) ----------
    {
        bf16* hw = sH[wave];
        #pragma unroll
        for (int t = 0; t < 8; ++t) {
            #pragma unroll
            for (int i = 0; i < 8; ++i) {
                float v = acc[t][i];
                v = v > 0.0f ? v : 0.0f;
                int mr = (lhalf << 3) + i;            // C layout: M = i + 8*lhalf
                hw[mr * N1 + t * 16 + lmod] = (bf16)v;
            }
        }
    }
    __syncthreads();

    // ---- GEMM2: z = h @ W2 (K = 128, 4 K-steps) ----------------------------
    v8f acc2[4];
    #pragma unroll
    for (int t = 0; t < 4; ++t)
        #pragma unroll
        for (int i = 0; i < 8; ++i) acc2[t][i] = 0.0f;

    const bf16* hw = sH[wave];
    #pragma unroll
    for (int k2 = 0; k2 < 4; ++k2) {
        // A fragment from the h strip: two 16-byte LDS vector loads per lane.
        v8bf lo = *(const v8bf*)&hw[lmod * N1 + k2 * 32 + koff];
        v8bf hi = *(const v8bf*)&hw[lmod * N1 + k2 * 32 + 16 + koff];
        v16bf a2 = __builtin_shufflevector(lo, hi,
            0, 1, 2, 3, 4, 5, 6, 7, 8, 9, 10, 11, 12, 13, 14, 15);

        v16bf bfrag2[4];
        #pragma unroll
        for (int t = 0; t < 4; ++t)
            bfrag2[t] = *(const v16bf*)&sW2[((k2 * 4 + t) * 32 + lane) * 16];
        #pragma unroll
        for (int t = 0; t < 4; ++t)
            acc2[t] = __builtin_amdgcn_wmma_f32_16x16x32_bf16(
                false, a2, false, bfrag2[t], (short)0, acc2[t], false, false);
    }

    // ---- store z strip (predicated on real rows) ---------------------------
    #pragma unroll
    for (int t = 0; t < 4; ++t) {
        #pragma unroll
        for (int i = 0; i < 8; ++i) {
            int r = rowBase + (lhalf << 3) + i;
            if (r < n_nodes)
                Z[(size_t)r * N2 + t * 16 + lmod] = acc2[t][i];
        }
    }
}

// 8 lanes per edge; z rows (256 B) stream from L2. out[0:E)=train, [E:2E)=false.
__global__ __launch_bounds__(256) void sgae_decode_kernel(
    const float* __restrict__ Z, const int* __restrict__ e_pos,
    const int* __restrict__ e_neg, float* __restrict__ out, int E)
{
    int gid = blockIdx.x * 32 + (threadIdx.x >> 3);
    if (gid >= 2 * E) return;
    int sub = threadIdx.x & 7;

    const int* ep = (gid < E) ? e_pos : e_neg;
    int e = (gid < E) ? gid : gid - E;
    int u = ep[2 * e + 0];
    int v = ep[2 * e + 1];

    const float4* zu = (const float4*)(Z + (size_t)u * N2);
    const float4* zv = (const float4*)(Z + (size_t)v * N2);
    float4 a0 = zu[sub],     b0 = zv[sub];
    float4 a1 = zu[sub + 8], b1 = zv[sub + 8];

    float s = a0.x * b0.x + a0.y * b0.y + a0.z * b0.z + a0.w * b0.w
            + a1.x * b1.x + a1.y * b1.y + a1.z * b1.z + a1.w * b1.w;
    s += __shfl_xor(s, 1, 8);
    s += __shfl_xor(s, 2, 8);
    s += __shfl_xor(s, 4, 8);

    if (sub == 0) out[gid] = 1.0f / (1.0f + __expf(-s));
}

extern "C" void kernel_launch(void* const* d_in, const int* in_sizes, int n_in,
                              void* d_out, int out_size, void* d_ws, size_t ws_size,
                              hipStream_t stream) {
    const float* X   = (const float*)d_in[0];   // [n_nodes, 512]
    const float* W   = (const float*)d_in[1];   // [512, 128]
    const float* W2  = (const float*)d_in[2];   // [128, 64]
    const int* e_pos = (const int*)d_in[3];     // [E, 2]
    const int* e_neg = (const int*)d_in[4];     // [E, 2]
    float* out = (float*)d_out;                 // [2E, 1]

    int n_nodes = in_sizes[0] / K1;
    int E       = in_sizes[3] / 2;
    float* Z    = (float*)d_ws;                 // [n_nodes, 64] f32 (25.6 MB)

    dim3 blk(256);
    sgae_encode_kernel<<<dim3((n_nodes + 127) / 128), blk, 0, stream>>>(
        X, W, W2, Z, n_nodes);
    sgae_decode_kernel<<<dim3((2 * E + 31) / 32), blk, 0, stream>>>(
        Z, e_pos, e_neg, out, E);
}
